// CascadeAttention_2d_11227044512047
// MI455X (gfx1250) — matrix operand order repaired and checked by reading, not compiled, not run on hardware
//
#include <hip/hip_runtime.h>

// ---------------------------------------------------------------------------
// CascadeAttention (EfficientViT CGA) for MI455X / gfx1250, wave32 + WMMA.
// Kernel A: fused cascade (QKV gemm -> BN -> dwconv -> attn -> softmax -> V*P)
//           one workgroup per batch element, everything resident in LDS.
// Kernel B: 512x512 projection GEMM + BN, WMMA-tiled through LDS with
//           global_load_async_to_lds_b128 stage-in.
// All matrices are staged in LDS in *fragment-major* order (per-lane 16-half
// chunks matching the v_wmma_f32_16x16x32_f16 VGPR layouts): fragment loads
// are ds_load_b128 pairs; score/k D-stores are single ds_store_b128.
// ---------------------------------------------------------------------------

typedef __attribute__((ext_vector_type(16))) _Float16 v16h;
typedef __attribute__((ext_vector_type(8)))  _Float16 v8h;
typedef __attribute__((ext_vector_type(8)))  float    v8f;

#define EPS_BN 1e-5f

#define NPIX 196      // 14*14
#define NPAD 208      // 13 * 16
#define KPAD 224      // 7 * 32 (attention K padding)

// ---- LDS layout for kernel A (bytes, all 16B aligned) ---------------------
#define SP32_OFF 0                        // float   [64][208]   cascade carry
#define SPH_OFF  53248                    // f16 frag B: 2kt x 13nt (qkv GEMM B)
#define WH_OFF   79872                    // f16 frag A: 6mt x 2kt  (qkv weights)
#define Q32_OFF  92160                    // float   [16][208]   q pre-dwconv
#define QB_OFF   105472                   // f16 frag B: 1kt x 13nt (q, scores B)
#define KT_OFF   118784                   // f16 frag A: 13mt x 1kt (k^T, scores A)
#define V16_OFF  132096                   // f16 frag A: 4mt x 7kt  (v, out A)
#define P_OFF    160768                   // f16 frag B: 7kt x 13nt (scores/probs)
#define QSC_OFF  253952                   // float[96] qkv BN scale
#define QSH_OFF  254336                   // float[96] qkv BN shift
#define DSC_OFF  254720                   // float[16] dw BN scale
#define DSH_OFF  254784                   // float[16] dw BN shift
#define AB_OFF   254848                   // float[196] attn bias table (head)
#define DWW_OFF  255632                   // float[16*49] dw weights
#define SMEM_A   258768

// ---------------------------------------------------------------------------
// Fragment-major addressing (units: halves). Each (tile,tile) block is
// 32 lanes x 16 halves = 512 halves; lane chunks are 32B and contiguous.
//
// A (16x32): lane = (m&15) | (k&16)
//            j    = k&15
__device__ __forceinline__ int aOff(int m, int k, int KT) {
  int kk = k & 31;
  int lane = (m & 15) | (kk & 16);
  int j = kk & 15;
  return ((((m >> 4) * KT) + (k >> 5)) << 9) + (lane << 4) + j;
}
// B (32x16): lane = (k&16) | (n&15), j = k&15
__device__ __forceinline__ int bOff(int k, int n, int NT) {
  int lane = (k & 16) | (n & 15);
  return ((((k >> 5) * NT) + (n >> 4)) << 9) + (lane << 4) + (k & 15);
}
__device__ __forceinline__ v16h fragA(const _Float16* mat, int mt, int kt,
                                      int KT, int lane) {
  return *(const v16h*)(mat + (((mt * KT) + kt) << 9) + (lane << 4));
}
__device__ __forceinline__ v16h fragB(const _Float16* mat, int kt, int nt,
                                      int NT, int lane) {
  return *(const v16h*)(mat + (((kt * NT) + nt) << 9) + (lane << 4));
}
__device__ __forceinline__ v8f wmma16(v16h a, v16h b, v8f c) {
  return __builtin_amdgcn_wmma_f32_16x16x32_f16(false, a, false, b,
                                                (short)0, c, false, false);
}
__device__ __forceinline__ float hswish(float v) {
  return v * fminf(fmaxf(v + 3.f, 0.f), 6.f) * (1.f / 6.f);
}

struct DwPtrs { const float* p[8]; };

// ===========================================================================
// Kernel A: fused cascade attention. grid = B(256) blocks, 256 thr (8 waves).
// ===========================================================================
__launch_bounds__(256, 1)
__global__ void cascade_attn_kernel(
    const float* __restrict__ x,
    const float* __restrict__ qkv_w,
    const float* __restrict__ qkv_g, const float* __restrict__ qkv_b,
    const float* __restrict__ qkv_m, const float* __restrict__ qkv_v,
    DwPtrs dww_g,
    const float* __restrict__ dw_g, const float* __restrict__ dw_b,
    const float* __restrict__ dw_m, const float* __restrict__ dw_v,
    const float* __restrict__ ab_g,
    const int* __restrict__ idx,
    _Float16* __restrict__ act)     // [B] x (16 ktile x 13 ntile) frag blocks
{
  extern __shared__ __align__(32) char smem[];
  float*    sp32 = (float*)   (smem + SP32_OFF);
  _Float16* spH  = (_Float16*)(smem + SPH_OFF);
  _Float16* wH   = (_Float16*)(smem + WH_OFF);
  float*    q32  = (float*)   (smem + Q32_OFF);
  _Float16* qB   = (_Float16*)(smem + QB_OFF);
  _Float16* kT   = (_Float16*)(smem + KT_OFF);
  _Float16* v16  = (_Float16*)(smem + V16_OFF);
  _Float16* P    = (_Float16*)(smem + P_OFF);
  float*    qsc  = (float*)   (smem + QSC_OFF);
  float*    qsh  = (float*)   (smem + QSH_OFF);
  float*    dsc  = (float*)   (smem + DSC_OFF);
  float*    dsh  = (float*)   (smem + DSH_OFF);
  float*    abL  = (float*)   (smem + AB_OFF);
  float*    dwwL = (float*)   (smem + DWW_OFF);

  const int tid  = threadIdx.x;
  const int wv   = tid >> 5;
  const int lane = tid & 31;
  const int b    = blockIdx.x;
  const int nn   = lane & 15;
  const int mb   = (lane & 16) ? 8 : 0;
  const _Float16 h0 = (_Float16)0.f;

  // ---- one-time zeroing of padded regions ----
  for (int p = tid; p < 64 * 12; p += 256)          // spH cols 196..207
    spH[bOff(p / 12, 196 + (p % 12), 13)] = h0;
  {
    v8h z = {};
    v8h* qz = (v8h*)qB;                              // qB: whole array
    v8h* kz = (v8h*)kT;                              // kT: whole array
    for (int p = tid; p < (13 * 512) / 8; p += 256) { qz[p] = z; kz[p] = z; }
  }
  for (int p = tid; p < 16 * NPAD; p += 256)        // P rows 208..223
    P[bOff(208 + p / NPAD, p % NPAD, 13)] = h0;
  __syncthreads();

  for (int i = 0; i < 8; ++i) {
    const int kd  = (i == 0) ? 7 : ((i == 1) ? 5 : 3);
    const int pad = kd >> 1;

    // ---- phase 0: per-head parameters + cascade input staging ----
    if (tid < 96) {
      float s = qkv_g[i * 96 + tid] * rsqrtf(qkv_v[i * 96 + tid] + EPS_BN);
      qsc[tid] = s;
      qsh[tid] = qkv_b[i * 96 + tid] - qkv_m[i * 96 + tid] * s;
    } else if (tid < 112) {
      int c = tid - 96;
      float s = dw_g[i * 16 + c] * rsqrtf(dw_v[i * 16 + c] + EPS_BN);
      dsc[c] = s;
      dsh[c] = dw_b[i * 16 + c] - dw_m[i * 16 + c] * s;
    }
    for (int p = tid; p < NPIX; p += 256) abL[p] = ab_g[i * NPIX + p];
    for (int p = tid; p < 16 * kd * kd; p += 256) dwwL[p] = dww_g.p[i][p];

    const float* wsrc = qkv_w + i * 96 * 64;
    for (int p = tid; p < 96 * 64; p += 256)
      wH[aOff(p >> 6, p & 63, 2)] = (_Float16)wsrc[p];

    const float* xh = x + ((size_t)b * 512 + (size_t)i * 64) * NPIX;
    for (int p = tid; p < 64 * NPIX; p += 256) {
      int c = p / NPIX, n = p % NPIX;
      float xv = xh[c * NPIX + n];
      float val = (i == 0) ? xv : (sp32[c * NPAD + n] + xv);
      spH[bOff(c, n, 13)] = (_Float16)val;
      if (i + 1 < 8) __builtin_prefetch(xh + 64 * NPIX + p, 0, 0);
    }
    __syncthreads();

    // ---- phase 1: QKV GEMM (96x196, K=64) + BN, split q/k/v ----
    // Straight-line loops; BN coefficients hoisted to registers per loop.
    {                                                // m-tile 0: q (f32)
      float sc[8], sh[8];
#pragma unroll
      for (int r = 0; r < 8; ++r) { sc[r] = qsc[mb + r]; sh[r] = qsh[mb + r]; }
      for (int nt = wv; nt < 13; nt += 8) {
        v8f acc = {};
#pragma unroll
        for (int kc = 0; kc < 2; ++kc)
          acc = wmma16(fragA(wH, 0, kc, 2, lane),
                       fragB(spH, kc, nt, 13, lane), acc);
        int n = nt * 16 + nn;
#pragma unroll
        for (int r = 0; r < 8; ++r)
          q32[(mb + r) * NPAD + n] = acc[r] * sc[r] + sh[r];
      }
    }
    {                                                // m-tile 1: k -> kT (A)
      float sc[8], sh[8];
#pragma unroll
      for (int r = 0; r < 8; ++r) {
        sc[r] = qsc[16 + mb + r];
        sh[r] = qsh[16 + mb + r];
      }
      for (int nt = wv; nt < 13; nt += 8) {
        v8f acc = {};
#pragma unroll
        for (int kc = 0; kc < 2; ++kc)
          acc = wmma16(fragA(wH, 1, kc, 2, lane),
                       fragB(spH, kc, nt, 13, lane), acc);
        v8h kv;
#pragma unroll
        for (int r = 0; r < 8; ++r)
          kv[r] = (_Float16)(acc[r] * sc[r] + sh[r]);
        // rows c=mb..mb+7 of column n: contiguous j of one kT lane chunk
        *(v8h*)(kT + (nt << 9) + (nn << 4) + mb) = kv;
      }
    }
#pragma unroll 1
    for (int mt = 2; mt < 6; ++mt) {                 // m-tiles 2..5: v (A)
      float sc[8], sh[8];
#pragma unroll
      for (int r = 0; r < 8; ++r) {
        sc[r] = qsc[mt * 16 + mb + r];
        sh[r] = qsh[mt * 16 + mb + r];
      }
      for (int nt = wv; nt < 13; nt += 8) {
        v8f acc = {};
#pragma unroll
        for (int kc = 0; kc < 2; ++kc)
          acc = wmma16(fragA(wH, mt, kc, 2, lane),
                       fragB(spH, kc, nt, 13, lane), acc);
        int n = nt * 16 + nn;
#pragma unroll
        for (int r = 0; r < 8; ++r)
          v16[aOff((mt - 2) * 16 + mb + r, n, 7)] =
              (_Float16)(acc[r] * sc[r] + sh[r]);
      }
    }
    __syncthreads();

    // ---- phase 2: depthwise conv on q + BN -> qB (f16, B-layout) ----
    for (int p = tid; p < 16 * NPIX; p += 256) {
      int c = p / NPIX, n = p % NPIX, yy = n / 14, xx = n % 14;
      float acc = 0.f;
      for (int dy = 0; dy < kd; ++dy) {
        int sy = yy + dy - pad;
        if (sy < 0 || sy >= 14) continue;
        for (int dx = 0; dx < kd; ++dx) {
          int sx = xx + dx - pad;
          if (sx < 0 || sx >= 14) continue;
          acc += dwwL[(c * kd + dy) * kd + dx] * q32[c * NPAD + sy * 14 + sx];
        }
      }
      qB[bOff(c, n, 13)] = (_Float16)(acc * dsc[c] + dsh[c]);
    }
    for (int p = tid; p < 64 * 28; p += 256)        // v K-cols 196..223 := 0
      v16[aOff(p / 28, 196 + (p % 28), 7)] = h0;
    __syncthreads();

    // ---- phase 3: score GEMM S' = k^T q (keys x queries, K=32 padded) ----
    // D rows = key m, D cols = query n; 8 row-values land on contiguous j of
    // one P lane chunk -> single 16B ds_store per lane.
    for (int t = wv; t < 13 * 13; t += 8) {
      int mtile = t / 13, ntile = t % 13;
      v8f acc = {};
      acc = wmma16(fragA(kT, mtile, 0, 1, lane),
                   fragB(qB, 0, ntile, 13, lane), acc);
      v8h sv;
#pragma unroll
      for (int r = 0; r < 8; ++r) sv[r] = (_Float16)acc[r];
      _Float16* dst = P + ((((mtile >> 1) * 13) + ntile) << 9)
                        + ((((mtile & 1) << 4) | nn) << 4) + mb;
      *(v8h*)dst = sv;
    }
    __syncthreads();

    // ---- phase 4: softmax over key m per query column n ----
    if (tid < NPIX) {
      int n = tid;
      const int* idxr = idx + n * NPIX;
      // per-32-row block base: P block (mb2, n>>4), this column's lane group
      int cbase = ((n >> 4) << 9) + ((n & 15) << 4);
      float mx = -1e30f;
#pragma unroll 1
      for (int mb2 = 0; mb2 < 7; ++mb2) {
        int lim = (mb2 == 6) ? 4 : 32;
        int base = (mb2 * 13 << 9) + cbase;
#pragma unroll 4
        for (int mm = 0; mm < lim; ++mm) {
          int o = base + ((mm & 16) << 4) + (mm & 15);
          float s = (float)P[o] * 0.25f + abL[idxr[mb2 * 32 + mm]];
          P[o] = (_Float16)s;
          mx = fmaxf(mx, s);
        }
      }
      float sum = 0.f;
#pragma unroll 1
      for (int mb2 = 0; mb2 < 7; ++mb2) {
        int lim = (mb2 == 6) ? 4 : 32;
        int base = (mb2 * 13 << 9) + cbase;
#pragma unroll 4
        for (int mm = 0; mm < lim; ++mm) {
          int o = base + ((mm & 16) << 4) + (mm & 15);
          float e = __expf((float)P[o] - mx);
          sum += e;
          P[o] = (_Float16)e;
        }
      }
      float rs = 1.f / sum;
#pragma unroll 1
      for (int mb2 = 0; mb2 < 7; ++mb2) {
        int lim = (mb2 == 6) ? 4 : 32;
        int base = (mb2 * 13 << 9) + cbase;
#pragma unroll 4
        for (int mm = 0; mm < lim; ++mm) {
          int o = base + ((mm & 16) << 4) + (mm & 15);
          P[o] = (_Float16)((float)P[o] * rs);
        }
      }
    } else {
      // zero score rows 196..207 (K-padding for the output GEMM)
      for (int p = tid - NPIX; p < 12 * NPAD; p += 60)
        P[bOff(196 + p / NPAD, p % NPAD, 13)] = h0;
    }
    __syncthreads();

    // ---- phase 5: output GEMM  sp = V (64x224) * P (224x208) ----
    for (int t = wv; t < 4 * 13; t += 8) {
      int dt = t / 13, nt = t % 13;
      v8f acc = {};
#pragma unroll
      for (int kc = 0; kc < 7; ++kc)
        acc = wmma16(fragA(v16, dt, kc, 7, lane),
                     fragB(P, kc, nt, 13, lane), acc);
      int n = nt * 16 + nn;
#pragma unroll
      for (int r = 0; r < 8; ++r)
        sp32[(dt * 16 + mb + r) * NPAD + n] = acc[r];
    }
    __syncthreads();

    // ---- phase 6: hard_swish -> fragment-major activation in d_ws ----
    // One 32B lane-chunk per loop step => coalesced 16B global stores.
    _Float16* actb = act + (size_t)b * 512 * NPAD;
#pragma unroll 1
    for (int q = tid; q < 2 * 13 * 32; q += 256) {
      int ktl = q / 416, r = q % 416, nt = r >> 5, ln = r & 31;
      int n = nt * 16 + (ln & 15);
      int dbase = ktl * 32 + (ln & 16);
      v8h lo, hi;
#pragma unroll
      for (int j = 0; j < 8; ++j) {
        float v0 = (n < NPIX) ? sp32[(dbase + j) * NPAD + n] : 0.f;
        float v1 = (n < NPIX) ? sp32[(dbase + 8 + j) * NPAD + n] : 0.f;
        lo[j] = (_Float16)hswish(v0);
        hi[j] = (_Float16)hswish(v1);
      }
      _Float16* gp = actb + (((i * 2 + ktl) * 13 + nt) << 9) + (ln << 4);
      *(v8h*)gp = lo;
      *((v8h*)gp + 1) = hi;
    }
    __syncthreads();
  }
}

// ===========================================================================
// Kernel B: projection GEMM out[b] = proj_w(512x512) * act[b] + BN.
// grid = (256 batches, 4 M-tiles of 128), 256 thr. Each wave: one 16-row
// M-tile x 13 N-tiles of f32 accumulators; K streamed in chunks of 64.
// Activation chunks arrive via global_load_async_to_lds_b128 (ASYNCcnt).
// ===========================================================================
#define WT_OFF_B 0          // f16 frag A: 8mt x 2kt  [128][64]
#define AC_OFF_B 16384      // f16 frag B: 2kt x 13nt [64][208]
#define PSC_OFF_B 43008
#define PSH_OFF_B 43520
#define SMEM_B   44032

__launch_bounds__(256, 1)
__global__ void proj_kernel(
    const _Float16* __restrict__ act,
    const float* __restrict__ pw,
    const float* __restrict__ pg, const float* __restrict__ pb,
    const float* __restrict__ pm, const float* __restrict__ pv,
    float* __restrict__ out)
{
  extern __shared__ __align__(32) char smem[];
  _Float16* wt  = (_Float16*)(smem + WT_OFF_B);
  _Float16* ac  = (_Float16*)(smem + AC_OFF_B);
  float*    psc = (float*)   (smem + PSC_OFF_B);
  float*    psh = (float*)   (smem + PSH_OFF_B);

  const int tid = threadIdx.x, wv = tid >> 5, lane = tid & 31;
  const int b = blockIdx.x;
  const int m0 = blockIdx.y * 128;

  if (tid < 128) {
    int o = m0 + tid;
    float s = pg[o] * rsqrtf(pv[o] + EPS_BN);
    psc[tid] = s;
    psh[tid] = pb[o] - pm[o] * s;
  }

  v8f acc[13];
#pragma unroll
  for (int j = 0; j < 13; ++j) acc[j] = (v8f){};

  const _Float16* actb = act + (size_t)b * 512 * NPAD;
#pragma unroll 1
  for (int kc = 0; kc < 8; ++kc) {
    __syncthreads();
    // stage weight tile [128][64] (f32 -> f16), fragment-major
    for (int p = tid; p < 128 * 64; p += 256) {
      int r = p >> 6, c = p & 63;
      wt[aOff(r, c, 2)] = (_Float16)pw[(size_t)(m0 + r) * 512 + kc * 64 + c];
    }
    // stage activation chunk (already fragment-major & contiguous in ws):
    // async DMA global -> LDS, 16B per lane, tracked by ASYNCcnt.
    for (int p = tid; p < (64 * NPAD) / 8; p += 256) {
      unsigned ldsoff = (unsigned)AC_OFF_B + ((unsigned)p << 4);
      unsigned goff = ((unsigned)(kc * 64 * NPAD) << 1) + ((unsigned)p << 4);
      asm volatile("global_load_async_to_lds_b128 %0, %1, %2 offset:0"
                   :: "v"(ldsoff), "v"(goff), "s"(actb)
                   : "memory");
    }
    asm volatile("s_wait_asynccnt 0" ::: "memory");
    __syncthreads();
#pragma unroll
    for (int ks = 0; ks < 2; ++ks) {
      v16h a = fragA(wt, wv, ks, 2, lane);
#pragma unroll
      for (int nt = 0; nt < 13; ++nt)
        acc[nt] = wmma16(a, fragB(ac, ks, nt, 13, lane), acc[nt]);
    }
  }

  const int nn = lane & 15, mb = (lane & 16) ? 8 : 0;
#pragma unroll
  for (int r = 0; r < 8; ++r) {
    int ml = wv * 16 + mb + r;
    int o  = m0 + ml;
    float s = psc[ml], sh = psh[ml];
#pragma unroll
    for (int nt = 0; nt < 13; ++nt) {
      int n = nt * 16 + nn;
      if (n < NPIX)
        out[((size_t)b * 512 + o) * NPIX + n] = acc[nt][r] * s + sh;
    }
  }
}

// ===========================================================================
extern "C" void kernel_launch(void* const* d_in, const int* in_sizes, int n_in,
                              void* d_out, int out_size, void* d_ws, size_t ws_size,
                              hipStream_t stream) {
  (void)in_sizes; (void)n_in; (void)out_size; (void)ws_size;

  const float* x      = (const float*)d_in[0];
  const float* qkv_w  = (const float*)d_in[1];
  const float* qkv_g  = (const float*)d_in[2];
  const float* qkv_b  = (const float*)d_in[3];
  const float* qkv_m  = (const float*)d_in[4];
  const float* qkv_v  = (const float*)d_in[5];
  DwPtrs dwp;
  for (int i = 0; i < 8; ++i) dwp.p[i] = (const float*)d_in[6 + i];
  const float* dw_g   = (const float*)d_in[14];
  const float* dw_b   = (const float*)d_in[15];
  const float* dw_m   = (const float*)d_in[16];
  const float* dw_v   = (const float*)d_in[17];
  const float* ab     = (const float*)d_in[18];
  const float* pw     = (const float*)d_in[19];
  const float* pg     = (const float*)d_in[20];
  const float* pb     = (const float*)d_in[21];
  const float* pm     = (const float*)d_in[22];
  const float* pv     = (const float*)d_in[23];
  const int*   idx    = (const int*)d_in[24];

  _Float16* actw = (_Float16*)d_ws;  // [256][512][208] f16 frag-major, 54.5 MB
  float*    out  = (float*)d_out;

  cascade_attn_kernel<<<dim3(256), dim3(256), SMEM_A, stream>>>(
      x, qkv_w, qkv_g, qkv_b, qkv_m, qkv_v, dwp,
      dw_g, dw_b, dw_m, dw_v, ab, idx, actw);

  proj_kernel<<<dim3(256, 4), dim3(256), SMEM_B, stream>>>(
      actw, pw, pg, pb, pm, pv, out);
}
